// MoETopKLayer_8546984919183
// MI455X (gfx1250) — matrix operand order repaired and checked by reading, not compiled, not run on hardware
//
#include <hip/hip_runtime.h>
#include <hip/hip_bf16.h>
#include <math.h>
#include <stdint.h>

// Problem sizes (fixed by the reference)
#define B_   32
#define S_   512
#define D_   512
#define E_   8
#define H_   2048
#define O_   512
#define TOPK 2

// Tiling
#define MT    64          // rows of x per workgroup (S tile)
#define HC    64          // hidden (H) chunk per pass
#define BLOCK 512         // 16 waves (wave32)

// LDS strides (elements), padded; all multiples of 8 -> 16B-aligned segments
#define XS   (D_ + 8)     // x_s   [MT][XS]
#define W1S  (D_ + 8)     // w1_s  [HC][W1S]   ([h][d], N-major for B frags)
#define W2S  (HC + 8)     // w2_s  [O_][W2S]   ([n][h], N-major for B frags)
#define HS   (HC + 8)     // h_s   [MT][HS]

typedef __attribute__((ext_vector_type(16))) __bf16 v16bf;
typedef __attribute__((ext_vector_type(8)))  __bf16 v8bf;
typedef __attribute__((ext_vector_type(8)))  float  v8f;

__device__ __forceinline__ float gelu_exact(float v) {
    return 0.5f * v * (1.0f + erff(v * 0.70710678118654752440f));
}

__device__ __forceinline__ v8f zero8() {
    v8f z = {0.f, 0.f, 0.f, 0.f, 0.f, 0.f, 0.f, 0.f};
    return z;
}

// Async global->LDS copy of one 16-byte segment (gfx1250 VFLAT async op,
// tracked by ASYNCcnt). vdst VGPR carries the LDS byte address.
__device__ __forceinline__ void async_ld_b128(uint32_t lds_byte_addr, const void* gaddr) {
    asm volatile("global_load_async_to_lds_b128 %0, %1, off"
                 :: "v"(lds_byte_addr), "v"(gaddr)
                 : "memory");
}

__device__ __forceinline__ void wait_async0() {
#if __has_builtin(__builtin_amdgcn_s_wait_asynccnt)
    __builtin_amdgcn_s_wait_asynccnt(0);
#else
    asm volatile("s_wait_asynccnt 0x0" ::: "memory");
#endif
}

// A fragment: 16x32 bf16 from row-major LDS [row][k].
// ISA 7.12.2: lanes 0-15 row M=lane, K = {0..7, 16..23}; lanes 16-31 K = {8..15, 24..31}.
__device__ __forceinline__ v16bf ld_afrag(const __bf16* base, int row0, int k0,
                                          int stride, int lane) {
    const __bf16* p = base + (row0 + (lane & 15)) * stride + k0 + ((lane >> 4) << 3);
    union { v16bf v; v8bf h[2]; } u;
    u.h[0] = *(const v8bf*)(p);
    u.h[1] = *(const v8bf*)(p + 16);
    return u.v;
}

// B fragment: 32x16 bf16 from N-major (transposed) LDS [n][k].
// lanes 0-15 col N=lane, K = {0..7, 16..23}; lanes 16-31 K = {8..15, 24..31}.
__device__ __forceinline__ v16bf ld_bfragT(const __bf16* baseT, int n0, int k0,
                                           int stride, int lane) {
    const __bf16* p = baseT + (n0 + (lane & 15)) * stride + k0 + ((lane >> 4) << 3);
    union { v16bf v; v8bf h[2]; } u;
    u.h[0] = *(const v8bf*)(p);
    u.h[1] = *(const v8bf*)(p + 16);
    return u.v;
}

__device__ __forceinline__ v8f wmma_bf16(v16bf a, v16bf b, v8f c) {
    // (neg_a, A, neg_b, B, c_mod, C, reuse_a, reuse_b)
    return __builtin_amdgcn_wmma_f32_16x16x32_bf16(false, a, false, b,
                                                   (short)0, c, false, false);
}

// ---------------------------------------------------------------------------
// Kernel 0: one-time fp32 -> bf16 weight convert + transpose into workspace.
//   w1t[e][h][d] = bf16(w1[e][d][h])   (B-frag friendly: D contiguous)
//   w2t[e][n][h] = bf16(w2[e][h][n])   (B-frag friendly: H contiguous)
// Both tensors have the same element count: E*D*H == E*H*O.
// ---------------------------------------------------------------------------
__global__ void __launch_bounds__(256, 1)
moe_cvt_kernel(const float* __restrict__ w1, const float* __restrict__ w2,
               __bf16* __restrict__ w1t, __bf16* __restrict__ w2t)
{
    const size_t N = (size_t)E_ * D_ * H_;
    const size_t stride = (size_t)gridDim.x * blockDim.x;
    for (size_t idx = (size_t)blockIdx.x * blockDim.x + threadIdx.x; idx < N; idx += stride) {
        {   // w1 transpose
            const size_t d = idx % D_;
            const size_t h = (idx / D_) % H_;
            const size_t e = idx / ((size_t)D_ * H_);
            w1t[idx] = (__bf16)w1[(e * D_ + d) * H_ + h];
        }
        {   // w2 transpose
            const size_t h = idx % H_;
            const size_t n = (idx / H_) % O_;
            const size_t e = idx / ((size_t)H_ * O_);
            w2t[idx] = (__bf16)w2[(e * H_ + h) * O_ + n];
        }
    }
}

// ---------------------------------------------------------------------------
// Kernel 1: attention pooling + gate softmax + top-2 routing (tiny).
// ---------------------------------------------------------------------------
__global__ void __launch_bounds__(256, 1)
moe_gate_kernel(const float* __restrict__ x,
                const float* __restrict__ attn_w, const float* __restrict__ attn_b,
                const float* __restrict__ gate_w, const float* __restrict__ gate_b,
                float* __restrict__ gbuf, int* __restrict__ ibuf)
{
    __shared__ float attw[D_];
    __shared__ float sc[S_];
    __shared__ float red[256];
    __shared__ float pooled[D_];
    __shared__ float gates[E_];

    const int t = threadIdx.x;
    const int b = blockIdx.x;

    for (int i = t; i < D_; i += 256) attw[i] = attn_w[i];
    __syncthreads();

    const float ab = attn_b[0];
    for (int s = t; s < S_; s += 256) {
        const float* xr = x + ((size_t)b * S_ + s) * D_;
        float acc = 0.f;
        for (int d = 0; d < D_; ++d) acc += xr[d] * attw[d];
        sc[s] = acc + ab;
    }
    __syncthreads();

    // softmax over S (each thread owns s = t and t+256)
    red[t] = fmaxf(sc[t], sc[t + 256]);
    __syncthreads();
    for (int off = 128; off > 0; off >>= 1) {
        if (t < off) red[t] = fmaxf(red[t], red[t + off]);
        __syncthreads();
    }
    const float mx = red[0];
    __syncthreads();
    float e0 = expf(sc[t] - mx), e1 = expf(sc[t + 256] - mx);
    sc[t] = e0; sc[t + 256] = e1;
    red[t] = e0 + e1;
    __syncthreads();
    for (int off = 128; off > 0; off >>= 1) {
        if (t < off) red[t] += red[t + off];
        __syncthreads();
    }
    const float inv = 1.0f / red[0];
    __syncthreads();

    // pooled = sum_s x * softmax
    for (int d = t; d < D_; d += 256) {
        float acc = 0.f;
        for (int s = 0; s < S_; ++s)
            acc += x[((size_t)b * S_ + s) * D_ + d] * sc[s];
        pooled[d] = acc * inv;
    }
    __syncthreads();

    if (t < E_) {
        float acc = gate_b[t];
        for (int d = 0; d < D_; ++d) acc += pooled[d] * gate_w[d * E_ + t];
        gates[t] = acc;
    }
    __syncthreads();

    if (t == 0) {
        float mg = gates[0];
        for (int e = 1; e < E_; ++e) mg = fmaxf(mg, gates[e]);
        float p[E_]; float s = 0.f;
        for (int e = 0; e < E_; ++e) { p[e] = expf(gates[e] - mg); s += p[e]; }
        for (int e = 0; e < E_; ++e) p[e] /= s;
        int i0 = 0;
        for (int e = 1; e < E_; ++e) if (p[e] > p[i0]) i0 = e;           // lowest idx on tie
        int i1 = (i0 == 0) ? 1 : 0;
        for (int e = 0; e < E_; ++e) if (e != i0 && p[e] > p[i1]) i1 = e;
        const float den = p[i0] + p[i1] + 1e-9f;
        gbuf[b * TOPK + 0] = p[i0] / den;
        gbuf[b * TOPK + 1] = p[i1] / den;
        ibuf[b * TOPK + 0] = i0;
        ibuf[b * TOPK + 1] = i1;
    }
}

// ---------------------------------------------------------------------------
// Kernel 2: fused routed expert FFN. WG = (batch b, 64-row S tile).
// Weight chunks staged with async global->LDS b128 copies (ASYNCcnt).
// ---------------------------------------------------------------------------
__global__ void __launch_bounds__(BLOCK, 1)
moe_ffn_kernel(const float* __restrict__ x,
               const __bf16* __restrict__ w1t, const float* __restrict__ b1,
               const __bf16* __restrict__ w2t, const float* __restrict__ b2,
               const float* __restrict__ gbuf, const int* __restrict__ ibuf,
               float* __restrict__ out)
{
    extern __shared__ __align__(16) char smem[];
    __bf16* x_s  = (__bf16*)smem;          // [MT][XS]
    __bf16* w1_s = x_s  + MT * XS;         // [HC][W1S] ([h][d])
    __bf16* w2_s = w1_s + HC * W1S;        // [O_][W2S] ([n][h])
    __bf16* h_s  = w2_s + O_ * W2S;        // [MT][HS]

    const int tid  = threadIdx.x;
    const int lane = tid & 31;
    const int wv   = tid >> 5;             // 0..15
    const int b    = blockIdx.y;
    const int s0   = blockIdx.x * MT;

    // Stage x tile (fp32 -> bf16) once; reused by both experts, all H chunks.
    for (int i = tid; i < MT * D_; i += BLOCK) {
        const int r = i >> 9, c = i & (D_ - 1);
        x_s[r * XS + c] = (__bf16)x[((size_t)b * S_ + s0 + r) * D_ + c];
    }

    const int rt    = wv & 3;                       // 16-row tile within MT
    const int cg    = wv >> 2;                      // column group 0..3
    const int mrow0 = rt * 16 + ((lane >> 4) << 3); // C-layout: rows m0..m0+7
    const int ncol  = lane & 15;                    // C-layout: column

    v8f oacc[8];
    #pragma unroll
    for (int t = 0; t < 8; ++t) oacc[t] = zero8();

    for (int kk = 0; kk < TOPK; ++kk) {
        const int   e = ibuf[b * TOPK + kk];
        const float g = gbuf[b * TOPK + kk];
        const __bf16* w1te = w1t + (size_t)e * H_ * D_;   // [h][d]
        const __bf16* w2te = w2t + (size_t)e * O_ * H_;   // [n][h]

        v8f c2[8];
        #pragma unroll
        for (int t = 0; t < 8; ++t) c2[t] = zero8();

        for (int hb = 0; hb < H_; hb += HC) {
            __syncthreads();   // previous chunk (and x staging) fully consumed

            // Async-stage w1 chunk: w1t[hb+hl][d] -> w1_s[hl][d], 16B segments.
            for (int j = tid; j < (HC * D_) / 8; j += BLOCK) {   // 4096 segs
                const int hl = j >> 6;             // j / (D_/8)
                const int d  = (j & 63) << 3;      // 8-elem block
                async_ld_b128((uint32_t)(uintptr_t)&w1_s[hl * W1S + d],
                              w1te + ((size_t)(hb + hl) * D_ + d));
            }
            // Async-stage w2 chunk: w2t[n][hb+hl] -> w2_s[n][hl], 16B segments.
            for (int j = tid; j < (O_ * HC) / 8; j += BLOCK) {   // 4096 segs
                const int n  = j >> 3;             // j / (HC/8)
                const int hl = (j & 7) << 3;
                async_ld_b128((uint32_t)(uintptr_t)&w2_s[n * W2S + hl],
                              w2te + ((size_t)n * H_ + hb + hl));
            }
            wait_async0();
            __syncthreads();

            // GEMM1: h = gelu(x_s[64x512] @ w1^T + b1) -> h_s[64xHC]
            {
                v8f c1 = zero8();
                #pragma unroll
                for (int k0 = 0; k0 < D_; k0 += 32) {
                    v16bf a  = ld_afrag(x_s, rt * 16, k0, XS, lane);
                    v16bf bb = ld_bfragT(w1_s, cg * 16, k0, W1S, lane);
                    c1 = wmma_bf16(a, bb, c1);
                }
                const int hcol   = cg * 16 + ncol;
                const float bias = b1[(size_t)e * H_ + hb + hcol];
                #pragma unroll
                for (int j = 0; j < 8; ++j) {
                    const float v = gelu_exact(c1[j] + bias);
                    h_s[(mrow0 + j) * HS + hcol] = (__bf16)v;
                }
            }
            __syncthreads();

            // GEMM2: c2 += h_s[64xHC] @ w2^T ; wave owns cols cg*128 .. cg*128+127
            {
                v16bf a0 = ld_afrag(h_s, rt * 16, 0,  HS, lane);
                v16bf a1 = ld_afrag(h_s, rt * 16, 32, HS, lane);
                #pragma unroll
                for (int t = 0; t < 8; ++t) {
                    const int n0 = cg * 128 + t * 16;
                    v16bf bb0 = ld_bfragT(w2_s, n0, 0,  W2S, lane);
                    c2[t] = wmma_bf16(a0, bb0, c2[t]);
                    v16bf bb1 = ld_bfragT(w2_s, n0, 32, W2S, lane);
                    c2[t] = wmma_bf16(a1, bb1, c2[t]);
                }
            }

            // Hint next weight chunk toward L2 (global_prefetch_b8)
            if (hb + HC < H_) {
                __builtin_prefetch(w1te + ((size_t)(hb + HC + (tid >> 3)) * D_), 0, 1);
                __builtin_prefetch(w2te + ((size_t)tid * H_ + hb + HC), 0, 1);
            }
        }

        // out_acc += g * gelu(c2 + b2)
        #pragma unroll
        for (int t = 0; t < 8; ++t) {
            const int n = cg * 128 + t * 16 + ncol;
            const float bias = b2[(size_t)e * O_ + n];
            #pragma unroll
            for (int j = 0; j < 8; ++j)
                oacc[t][j] += g * gelu_exact(c2[t][j] + bias);
        }
    }

    // Single store of the finished tile (no atomics, out fully overwritten)
    #pragma unroll
    for (int t = 0; t < 8; ++t) {
        const int n = cg * 128 + t * 16 + ncol;
        #pragma unroll
        for (int j = 0; j < 8; ++j)
            out[((size_t)b * S_ + s0 + mrow0 + j) * O_ + n] = oacc[t][j];
    }
}

// ---------------------------------------------------------------------------
extern "C" void kernel_launch(void* const* d_in, const int* in_sizes, int n_in,
                              void* d_out, int out_size, void* d_ws, size_t ws_size,
                              hipStream_t stream) {
    (void)in_sizes; (void)n_in; (void)out_size; (void)ws_size;

    const float* x      = (const float*)d_in[0];
    const float* attn_w = (const float*)d_in[1];
    const float* attn_b = (const float*)d_in[2];
    const float* gate_w = (const float*)d_in[3];
    const float* gate_b = (const float*)d_in[4];
    const float* w1     = (const float*)d_in[5];
    const float* b1     = (const float*)d_in[6];
    const float* w2     = (const float*)d_in[7];
    const float* b2     = (const float*)d_in[8];
    float* out = (float*)d_out;

    // Workspace layout
    float*  gbuf = (float*)d_ws;                                  // 256 B
    int*    ibuf = (int*)((char*)d_ws + 256);                     // 256 B
    __bf16* w1t  = (__bf16*)((char*)d_ws + 4096);                 // 16 MB
    __bf16* w2t  = w1t + (size_t)E_ * H_ * D_;                    // 16 MB

    // One-time (per launch) weight convert + transpose to bf16.
    moe_cvt_kernel<<<8192, 256, 0, stream>>>(w1, w2, w1t, w2t);

    moe_gate_kernel<<<B_, 256, 0, stream>>>(x, attn_w, attn_b, gate_w, gate_b,
                                            gbuf, ibuf);

    const size_t smem_bytes =
        (size_t)(MT * XS + HC * W1S + O_ * W2S + MT * HS) * sizeof(__bf16); // ~216 KB
    hipFuncSetAttribute((const void*)moe_ffn_kernel,
                        hipFuncAttributeMaxDynamicSharedMemorySize,
                        (int)smem_bytes);

    dim3 grid(S_ / MT, B_);   // 8 x 32 = 256 workgroups
    moe_ffn_kernel<<<grid, BLOCK, smem_bytes, stream>>>(x, w1t, b1, w2t, b2,
                                                        gbuf, ibuf, out);
}
